// DeformableTransformer2D_25924422599100
// MI455X (gfx1250) — matrix operand order repaired and checked by reading, not compiled, not run on hardware
//
#include <hip/hip_runtime.h>
#include <hip/hip_bf16.h>
#include <math.h>

// ---------------------------------------------------------------------------
// Types for CDNA5 WMMA (wave32)
// ---------------------------------------------------------------------------
typedef __attribute__((ext_vector_type(16))) __bf16 bf16x16;
typedef __attribute__((ext_vector_type(8)))  float  f32x8;

__device__ __forceinline__ __bf16 f32_to_bf16(float f) {
    union { float f; unsigned int u; } v; v.f = f;
    unsigned int r = v.u + 0x7FFFu + ((v.u >> 16) & 1u);   // round-to-nearest-even
    unsigned short h = (unsigned short)(r >> 16);
    return __builtin_bit_cast(__bf16, h);
}

// ---------------------------------------------------------------------------
// Problem constants
// ---------------------------------------------------------------------------
#define CB   4
#define CC   256
#define CH   100
#define CW   100
#define CNQ  (CH * CW)
#define CM   (CB * CNQ)
#define CDFF 1024
#define CNH  8
#define CNP  4

// GEMM tiling: block = 128x64, 8 waves, each wave 32x32 (2x2 WMMA 16x16x32)
#define BM 128
#define BN 64
#define KT 32
#define LDS_STRIDE 48   // 32 + 16 pad, keeps 16B alignment per chunk

// ---------------------------------------------------------------------------
// 1) Transpose (b,c,nq) -> (b,nq,c): q as f32 (residual), v as bf16 (GEMM A)
// ---------------------------------------------------------------------------
__global__ __launch_bounds__(256)
void prep_qv_kernel(const float* __restrict__ query, const float* __restrict__ value,
                    float* __restrict__ q, __bf16* __restrict__ vbf)
{
    __shared__ float tq[32][33];
    __shared__ float tv[32][33];
    const int b  = blockIdx.z;
    const int p0 = blockIdx.x * 32;
    const int c0 = blockIdx.y * 32;
    const int tx = threadIdx.x & 31;
    const int ty = threadIdx.x >> 5;         // 0..7
    const float* qb = query + (size_t)b * CC * CNQ;
    const float* vb = value + (size_t)b * CC * CNQ;
    #pragma unroll
    for (int i = 0; i < 32; i += 8) {
        int ch  = c0 + ty + i;
        int pix = p0 + tx;
        float qv = 0.f, vv = 0.f;
        if (pix < CNQ) {
            qv = qb[(size_t)ch * CNQ + pix];
            vv = vb[(size_t)ch * CNQ + pix];
        }
        tq[ty + i][tx] = qv;
        tv[ty + i][tx] = vv;
    }
    __syncthreads();
    float*  qo = q   + (size_t)b * CNQ * CC;
    __bf16* vo = vbf + (size_t)b * CNQ * CC;
    #pragma unroll
    for (int i = 0; i < 32; i += 8) {
        int pix = p0 + ty + i;
        int ch  = c0 + tx;
        if (pix < CNQ) {
            qo[(size_t)pix * CC + ch] = tq[tx][ty + i];
            vo[(size_t)pix * CC + ch] = f32_to_bf16(tv[tx][ty + i]);
        }
    }
}

// ---------------------------------------------------------------------------
// 2) LayerNorm over C=256 (one block per row), optional +pos, bf16 output
// ---------------------------------------------------------------------------
template<bool ADD_POS>
__global__ __launch_bounds__(256)
void layernorm_kernel(const float* __restrict__ x, const float* __restrict__ pos,
                      const float* __restrict__ g, const float* __restrict__ bta,
                      __bf16* __restrict__ out)
{
    __shared__ float red[8];
    __shared__ float stat[2];
    const int row  = blockIdx.x;
    const int tid  = threadIdx.x;
    const int lane = tid & 31;
    const int wid  = tid >> 5;
    const float v = x[(size_t)row * CC + tid];

    float s = v;
    #pragma unroll
    for (int o = 16; o > 0; o >>= 1) s += __shfl_down(s, o, 32);
    if (lane == 0) red[wid] = s;
    __syncthreads();
    if (tid == 0) {
        float t = 0.f;
        #pragma unroll
        for (int i = 0; i < 8; ++i) t += red[i];
        stat[0] = t * (1.0f / CC);
    }
    __syncthreads();
    const float mean = stat[0];
    const float d = v - mean;
    float s2 = d * d;
    #pragma unroll
    for (int o = 16; o > 0; o >>= 1) s2 += __shfl_down(s2, o, 32);
    if (lane == 0) red[wid] = s2;
    __syncthreads();
    if (tid == 0) {
        float t = 0.f;
        #pragma unroll
        for (int i = 0; i < 8; ++i) t += red[i];
        stat[1] = t * (1.0f / CC);
    }
    __syncthreads();
    float y = d * rsqrtf(stat[1] + 1e-5f) * g[tid] + bta[tid];
    if (ADD_POS) y += pos[(size_t)row * CC + tid];
    out[(size_t)row * CC + tid] = f32_to_bf16(y);
}

// ---------------------------------------------------------------------------
// 3) f32 -> bf16 weight conversion
// ---------------------------------------------------------------------------
__global__ __launch_bounds__(256)
void conv_bf16_kernel(const float* __restrict__ in, __bf16* __restrict__ out, int n)
{
    int i = blockIdx.x * 256 + threadIdx.x;
    if (i < n) out[i] = f32_to_bf16(in[i]);
}

// ---------------------------------------------------------------------------
// 4) WMMA bf16 GEMM:  C[M,N] = A[M,K] @ W[N,K]^T (+bias) (+epilogue)
//    Double-buffered LDS fed by branchless GLOBAL_LOAD_ASYNC_TO_LDS_B128
//    (ASYNCcnt). Source rows are CLAMPED (not skipped) so issuance has no
//    divergent control flow; duplicated tile rows/cols only feed outputs the
//    guarded epilogue discards.
//    EPI 0: store f32
//    EPI 1: store f32 + residual
//    EPI 2: exact GELU, store bf16
//    EPI 3: residual, store f32 transposed to (b, N, rows_per_batch)
// ---------------------------------------------------------------------------
template<int EPI>
__global__ __launch_bounds__(256)
void gemm_bf16_kernel(const __bf16* __restrict__ A, const __bf16* __restrict__ Wt,
                      const float* __restrict__ bias, const float* __restrict__ resid,
                      float* __restrict__ outF, __bf16* __restrict__ outB,
                      int M, int N, int K, int rows_per_batch)
{
    __shared__ __attribute__((aligned(16))) __bf16 As[2][BM * LDS_STRIDE];
    __shared__ __attribute__((aligned(16))) __bf16 Bs[2][BN * LDS_STRIDE];

    const int tid  = threadIdx.x;
    const int lane = tid & 31;
    const int wave = tid >> 5;
    const int wm   = wave & 3;      // wave row   -> M offset wm*32
    const int wn   = wave >> 2;     // wave col   -> N offset wn*32
    const int m0   = blockIdx.x * BM;
    const int n0   = blockIdx.y * BN;
    const int halfsel = lane >> 4;  // 0: lanes 0-15, 1: lanes 16-31
    const int l16     = lane & 15;

    // per-thread staging assignment: chunks of 8 bf16 (16 bytes)
    const int arow0 = tid >> 2;                 // 0..63
    const int ako0  = (tid & 3) << 3;
    const int arow1 = (tid + 256) >> 2;         // 64..127
    const int ako1  = ((tid + 256) & 3) << 3;
    const int brow  = tid >> 2;                 // 0..63
    const int bko   = (tid & 3) << 3;
    // clamp source rows -> always-in-bounds reads, branchless issuance
    const int ar0 = min(m0 + arow0, M - 1);
    const int ar1 = min(m0 + arow1, M - 1);
    const int br  = min(n0 + brow,  N - 1);
    const __bf16* aptr0 = A  + (size_t)ar0 * K + ako0;
    const __bf16* aptr1 = A  + (size_t)ar1 * K + ako1;
    const __bf16* bptr  = Wt + (size_t)br  * K + bko;

    // issue one tile's async copies into LDS buffer `bufi` (no branches)
    auto issue_tile = [&](int bufi, int k0) {
        unsigned la0 = (unsigned)(size_t)(&As[bufi][arow0 * LDS_STRIDE + ako0]);
        unsigned la1 = (unsigned)(size_t)(&As[bufi][arow1 * LDS_STRIDE + ako1]);
        unsigned lb  = (unsigned)(size_t)(&Bs[bufi][brow  * LDS_STRIDE + bko]);
        asm volatile("global_load_async_to_lds_b128 %0, %1, off"
                     :: "v"(la0), "v"(aptr0 + k0) : "memory");
        asm volatile("global_load_async_to_lds_b128 %0, %1, off"
                     :: "v"(la1), "v"(aptr1 + k0) : "memory");
        asm volatile("global_load_async_to_lds_b128 %0, %1, off"
                     :: "v"(lb), "v"(bptr + k0) : "memory");
    };

    f32x8 acc[2][2] = {};
    const int nk = K / KT;      // K is a multiple of 32 here (256 or 1024)

    issue_tile(0, 0);
    for (int kt = 0; kt < nk; ++kt) {
        const int bufi = kt & 1;
        asm volatile("s_wait_asynccnt 0x0" ::: "memory");   // own tile landed
        __syncthreads();                                    // everyone's tile landed;
                                                            // other buffer fully consumed
        if (kt + 1 < nk) issue_tile(bufi ^ 1, (kt + 1) * KT);

        // ---- fragments per ISA layouts (wave32) ----
        bf16x16 afrag[2], bfrag[2];
        #pragma unroll
        for (int i = 0; i < 2; ++i) {
            int arow = wm * 32 + i * 16 + l16;     // A row = M index
            union { uint4 q[2]; bf16x16 v; } u;
            // lane<16: K 0-7 then 16-23 ; lane>=16: K 8-15 then 24-31
            u.q[0] = *reinterpret_cast<const uint4*>(&As[bufi][arow * LDS_STRIDE + halfsel * 8]);
            u.q[1] = *reinterpret_cast<const uint4*>(&As[bufi][arow * LDS_STRIDE + 16 + halfsel * 8]);
            afrag[i] = u.v;
        }
        #pragma unroll
        for (int j = 0; j < 2; ++j) {
            int brw = wn * 32 + j * 16 + l16;      // B lane = N column
            union { uint4 q[2]; bf16x16 v; } u;
            // lane<16: K 0-7 then 16-23 ; lane>=16: K 8-15 then 24-31
            u.q[0] = *reinterpret_cast<const uint4*>(&Bs[bufi][brw * LDS_STRIDE + halfsel * 8]);
            u.q[1] = *reinterpret_cast<const uint4*>(&Bs[bufi][brw * LDS_STRIDE + 16 + halfsel * 8]);
            bfrag[j] = u.v;
        }
        #pragma unroll
        for (int i = 0; i < 2; ++i)
            #pragma unroll
            for (int j = 0; j < 2; ++j)
                acc[i][j] = __builtin_amdgcn_wmma_f32_16x16x32_bf16(
                    false, afrag[i], false, bfrag[j], (short)0, acc[i][j], false, false);
        // no trailing barrier: next iteration's wait+barrier protects the buffers
    }

    // ---- epilogue (C/D layout: VGPR r = row r, +8 for upper half-wave) ----
    #pragma unroll
    for (int i = 0; i < 2; ++i) {
        #pragma unroll
        for (int j = 0; j < 2; ++j) {
            int col = n0 + wn * 32 + j * 16 + l16;
            if (col >= N) continue;
            int rowBase = m0 + wm * 32 + i * 16 + halfsel * 8;
            float bv = bias ? bias[col] : 0.f;
            #pragma unroll
            for (int r = 0; r < 8; ++r) {
                int row = rowBase + r;
                if (row >= M) continue;
                float v = acc[i][j][r] + bv;
                size_t idx = (size_t)row * N + col;
                if (EPI == 0) {
                    outF[idx] = v;
                } else if (EPI == 1) {
                    outF[idx] = v + resid[idx];
                } else if (EPI == 2) {
                    float gg = 0.5f * v * (1.0f + erff(v * 0.70710678118654752f));
                    outB[idx] = f32_to_bf16(gg);
                } else {
                    float o = v + resid[idx];
                    int bb  = row / rows_per_batch;
                    int pix = row - bb * rows_per_batch;
                    outF[((size_t)bb * N + col) * rows_per_batch + pix] = o;
                }
            }
        }
    }
}

// ---------------------------------------------------------------------------
// 5) Deformable attention sampling: one block per (b,q); thread = (head, ch)
// ---------------------------------------------------------------------------
__global__ __launch_bounds__(256)
void deform_attn_kernel(const float* __restrict__ off, const float* __restrict__ logits,
                        const float* __restrict__ vp, __bf16* __restrict__ attn)
{
    const int bq = blockIdx.x;            // 0 .. B*NQ-1
    const int b  = bq / CNQ;
    const int qi = bq - b * CNQ;
    const int h  = threadIdx.x >> 5;      // head 0..7
    const int d  = threadIdx.x & 31;      // channel within head

    const float* o  = off    + (size_t)bq * (CNH * CNP * 2) + h * (CNP * 2);
    const float* lg = logits + (size_t)bq * (CNH * CNP)     + h * CNP;

    // softmax over the 4 points (redundant per lane; cheap)
    float l0 = lg[0], l1 = lg[1], l2 = lg[2], l3 = lg[3];
    float mx = fmaxf(fmaxf(l0, l1), fmaxf(l2, l3));
    float e0 = expf(l0 - mx), e1 = expf(l1 - mx), e2 = expf(l2 - mx), e3 = expf(l3 - mx);
    float inv = 1.0f / (e0 + e1 + e2 + e3);
    float wsm[4] = { e0 * inv, e1 * inv, e2 * inv, e3 * inv };

    // reference-point quirk replicated exactly: ref0 = (i/h), ref1 = (j/w)
    const float ref0 = (float)(qi / CW) * (1.0f / CH);
    const float ref1 = (float)(qi % CW) * (1.0f / CW);

    const float* vpb = vp + ((size_t)b * CNQ) * CC + (h * 32 + d);

    float res = 0.f;
    #pragma unroll
    for (int p = 0; p < CNP; ++p) {
        float ox = o[p * 2 + 0], oy = o[p * 2 + 1];
        float gx = (ref0 + ox * (1.0f / CW)) * CW - 0.5f;
        float gy = (ref1 + oy * (1.0f / CH)) * CH - 0.5f;
        float x0f = floorf(gx), y0f = floorf(gy);
        int   x0 = (int)x0f,    y0 = (int)y0f;
        int   x1 = x0 + 1,      y1 = y0 + 1;
        float wx1 = gx - x0f, wx0 = 1.0f - wx1;
        float wy1 = gy - y0f, wy0 = 1.0f - wy1;

        float s00 = 0.f, s10 = 0.f, s01 = 0.f, s11 = 0.f;
        if (x0 >= 0 && x0 < CW && y0 >= 0 && y0 < CH) s00 = vpb[(size_t)(y0 * CW + x0) * CC];
        if (x1 >= 0 && x1 < CW && y0 >= 0 && y0 < CH) s10 = vpb[(size_t)(y0 * CW + x1) * CC];
        if (x0 >= 0 && x0 < CW && y1 >= 0 && y1 < CH) s01 = vpb[(size_t)(y1 * CW + x0) * CC];
        if (x1 >= 0 && x1 < CW && y1 >= 0 && y1 < CH) s11 = vpb[(size_t)(y1 * CW + x1) * CC];

        float samp = s00 * (wx0 * wy0) + s10 * (wx1 * wy0)
                   + s01 * (wx0 * wy1) + s11 * (wx1 * wy1);
        res += wsm[p] * samp;
    }
    attn[(size_t)bq * CC + h * 32 + d] = f32_to_bf16(res);
}

// ---------------------------------------------------------------------------
// Launch
// ---------------------------------------------------------------------------
extern "C" void kernel_launch(void* const* d_in, const int* in_sizes, int n_in,
                              void* d_out, int out_size, void* d_ws, size_t ws_size,
                              hipStream_t stream) {
    (void)in_sizes; (void)n_in; (void)out_size; (void)ws_size;

    const float* query     = (const float*)d_in[0];
    const float* value     = (const float*)d_in[1];
    const float* query_pos = (const float*)d_in[2];
    const float* ln1_g     = (const float*)d_in[3];
    const float* ln1_b     = (const float*)d_in[4];
    const float* W_off     = (const float*)d_in[5];
    const float* b_off     = (const float*)d_in[6];
    const float* W_attn    = (const float*)d_in[7];
    const float* b_attn    = (const float*)d_in[8];
    const float* W_v       = (const float*)d_in[9];
    const float* b_v       = (const float*)d_in[10];
    const float* W_out     = (const float*)d_in[11];
    const float* b_out     = (const float*)d_in[12];
    const float* ln2_g     = (const float*)d_in[13];
    const float* ln2_b     = (const float*)d_in[14];
    const float* W_f1      = (const float*)d_in[15];
    const float* b_f1      = (const float*)d_in[16];
    const float* W_f2      = (const float*)d_in[17];
    const float* b_f2      = (const float*)d_in[18];
    float* out = (float*)d_out;

    // workspace carve-up (256B aligned regions)
    char* ws = (char*)d_ws;
    size_t cur = 0;
    auto alloc = [&](size_t bytes) -> void* {
        void* p = ws + cur;
        cur += (bytes + 255) & ~(size_t)255;
        return p;
    };
    float*  q_f32   = (float*) alloc((size_t)CM * CC * 4);
    float*  q2_f32  = (float*) alloc((size_t)CM * CC * 4);
    float*  vp_f32  = (float*) alloc((size_t)CM * CC * 4);
    float*  off_f32 = (float*) alloc((size_t)CM * 64 * 4);
    float*  lg_f32  = (float*) alloc((size_t)CM * 32 * 4);
    __bf16* v_bf    = (__bf16*)alloc((size_t)CM * CC * 2);
    __bf16* qp_bf   = (__bf16*)alloc((size_t)CM * CC * 2);
    __bf16* attn_bf = (__bf16*)alloc((size_t)CM * CC * 2);
    __bf16* ln2_bf  = (__bf16*)alloc((size_t)CM * CC * 2);
    __bf16* f1_bf   = (__bf16*)alloc((size_t)CM * CDFF * 2);
    __bf16* wv_bf   = (__bf16*)alloc((size_t)CC * CC * 2);
    __bf16* woff_bf = (__bf16*)alloc((size_t)64 * CC * 2);
    __bf16* wattn_bf= (__bf16*)alloc((size_t)32 * CC * 2);
    __bf16* wout_bf = (__bf16*)alloc((size_t)CC * CC * 2);
    __bf16* wf1_bf  = (__bf16*)alloc((size_t)CDFF * CC * 2);
    __bf16* wf2_bf  = (__bf16*)alloc((size_t)CC * CDFF * 2);

    // weight conversions f32 -> bf16
    auto conv = [&](const float* src, __bf16* dst, int n) {
        conv_bf16_kernel<<<(n + 255) / 256, 256, 0, stream>>>(src, dst, n);
    };
    conv(W_v,    wv_bf,    CC * CC);
    conv(W_off,  woff_bf,  64 * CC);
    conv(W_attn, wattn_bf, 32 * CC);
    conv(W_out,  wout_bf,  CC * CC);
    conv(W_f1,   wf1_bf,   CDFF * CC);
    conv(W_f2,   wf2_bf,   CC * CDFF);

    // 1) transpose q/v into (b,nq,c)
    dim3 gPrep((CNQ + 31) / 32, CC / 32, CB);
    prep_qv_kernel<<<gPrep, 256, 0, stream>>>(query, value, q_f32, v_bf);

    // 2) LN1 + pos -> qp (bf16)
    layernorm_kernel<true><<<CM, 256, 0, stream>>>(q_f32, query_pos, ln1_g, ln1_b, qp_bf);

    // 3) value projection: vp = v @ W_v^T + b_v  (f32 out for sampling)
    dim3 gV((CM + BM - 1) / BM, (CC + BN - 1) / BN);
    gemm_bf16_kernel<0><<<gV, 256, 0, stream>>>(v_bf, wv_bf, b_v, nullptr,
                                                vp_f32, nullptr, CM, CC, CC, CNQ);
    // 4) offsets: (M,64) and attention logits: (M,32)
    dim3 gO((CM + BM - 1) / BM, 1);
    gemm_bf16_kernel<0><<<gO, 256, 0, stream>>>(qp_bf, woff_bf, b_off, nullptr,
                                                off_f32, nullptr, CM, 64, CC, CNQ);
    gemm_bf16_kernel<0><<<gO, 256, 0, stream>>>(qp_bf, wattn_bf, b_attn, nullptr,
                                                lg_f32, nullptr, CM, 32, CC, CNQ);

    // 5) softmax + bilinear sampling + head-combine -> attn (bf16)
    deform_attn_kernel<<<CM, 256, 0, stream>>>(off_f32, lg_f32, vp_f32, attn_bf);

    // 6) output projection + residual: q2 = q + attn @ W_out^T + b_out
    gemm_bf16_kernel<1><<<gV, 256, 0, stream>>>(attn_bf, wout_bf, b_out, q_f32,
                                                q2_f32, nullptr, CM, CC, CC, CNQ);

    // 7) LN2 -> bf16
    layernorm_kernel<false><<<CM, 256, 0, stream>>>(q2_f32, nullptr, ln2_g, ln2_b, ln2_bf);

    // 8) FFN1 + exact GELU -> f1 (bf16)
    dim3 gF1((CM + BM - 1) / BM, (CDFF + BN - 1) / BN);
    gemm_bf16_kernel<2><<<gF1, 256, 0, stream>>>(ln2_bf, wf1_bf, b_f1, nullptr,
                                                 nullptr, f1_bf, CM, CDFF, CC, CNQ);

    // 9) FFN2 + residual + transposed store straight into (b,c,h,w) output
    gemm_bf16_kernel<3><<<gV, 256, 0, stream>>>(f1_bf, wf2_bf, b_f2, q2_f32,
                                                out, nullptr, CM, CC, CDFF, CNQ);
}
